// NEXT_Simulator_28321014349932
// MI455X (gfx1250) — matrix operand order repaired and checked by reading, not compiled, not run on hardware
//
#include <hip/hip_runtime.h>
#include <hip/hip_bf16.h>
#include <cstdint>

// ---------------------------------------------------------------------------
// NEXT detector simulator for MI455X (gfx1250, wave32, WMMA).
// Dominant cost: sipm einsum 'bnt,bns->bst' == per-batch GEMM
//   C[s,t] = sum_n R[n,s] * E[n,t],  S=2209, T=550, K=5000, B=4  (~48.6 GFLOP)
// v_wmma_f32_16x16x32_f16 with:
//   - A (R^T) generated in registers, reused across 8 t-tiles per wave
//   - B (E)  staged in LDS as packed 16x16 f16 sub-tiles, loaded with
//     ds_load_b128; loads + s_wait_dscnt fused in ONE asm block so the
//     scheduler cannot place a WMMA between a load and its wait.
// ---------------------------------------------------------------------------

#define BATCH      4
#define N_ELEC     5000
#define N_TICKS    550
#define N_SIPM     47
#define S_TOT      (N_SIPM * N_SIPM)      // 2209
#define GAUSS_NORM 0.8920620581f          // 1/(sqrt(0.2)*sqrt(2*pi))
#define INV_2SIG   2.5f                   // 1/(2*0.2)

typedef __attribute__((ext_vector_type(16))) _Float16 v16h;
typedef __attribute__((ext_vector_type(8)))  _Float16 v8h;
typedef __attribute__((ext_vector_type(2)))  _Float16 h2;
typedef __attribute__((ext_vector_type(8)))  float    v8f;

__device__ __forceinline__ float sigmoidf(float v) {
    return 1.0f / (1.0f + __expf(-v));
}

// ---------------------------------------------------------------------------
// Kernel 1: per-electron MLPs.
//   pmt_resp[b,n,12] (f16, ws)  = sigmoid-MLP 2->28->28->28->12
//   amp[b,n]         (f32, ws)  = sipm_norm * (1 + 0.5*tanh(MLP 2->28->1))
// ---------------------------------------------------------------------------
__global__ __launch_bounds__(256) void mlp_kernel(
    const float* __restrict__ electrons,
    const float* __restrict__ pmt_w1, const float* __restrict__ pmt_b1,
    const float* __restrict__ pmt_w2, const float* __restrict__ pmt_b2,
    const float* __restrict__ pmt_w3, const float* __restrict__ pmt_b3,
    const float* __restrict__ pmt_w4, const float* __restrict__ pmt_b4,
    const float* __restrict__ si_w1,  const float* __restrict__ si_b1,
    const float* __restrict__ si_w2,  const float* __restrict__ si_b2,
    const float* __restrict__ sipm_norm,
    _Float16* __restrict__ pmt_ws, float* __restrict__ amp_ws)
{
    const int gid = blockIdx.x * 256 + threadIdx.x;
    if (gid >= BATCH * N_ELEC) return;

    const float x = electrons[(size_t)gid * 3 + 0];
    const float y = electrons[(size_t)gid * 3 + 1];

    float h1[28], h2a[28], h3[28];
    #pragma unroll
    for (int j = 0; j < 28; ++j)
        h1[j] = sigmoidf(x * pmt_w1[j] + y * pmt_w1[28 + j] + pmt_b1[j]);
    for (int j = 0; j < 28; ++j) {
        float s = pmt_b2[j];
        #pragma unroll
        for (int k = 0; k < 28; ++k) s += h1[k] * pmt_w2[k * 28 + j];
        h2a[j] = sigmoidf(s);
    }
    for (int j = 0; j < 28; ++j) {
        float s = pmt_b3[j];
        #pragma unroll
        for (int k = 0; k < 28; ++k) s += h2a[k] * pmt_w3[k * 28 + j];
        h3[j] = sigmoidf(s);
    }
    #pragma unroll
    for (int j = 0; j < 12; ++j) {
        float s = pmt_b4[j];
        #pragma unroll
        for (int k = 0; k < 28; ++k) s += h3[k] * pmt_w4[k * 12 + j];
        pmt_ws[(size_t)gid * 12 + j] = (_Float16)sigmoidf(s);
    }

    float t = si_b2[0];
    #pragma unroll
    for (int j = 0; j < 28; ++j) {
        float a = sigmoidf(x * si_w1[j] + y * si_w1[28 + j] + si_b1[j]);
        t += a * si_w2[j];
    }
    amp_ws[gid] = sipm_norm[0] * (1.0f + 0.5f * tanhf(t));
}

// ---------------------------------------------------------------------------
// sE layout: 16 sub-tiles (tt=0..7 tick-tiles x nh=0..1 K-halves),
// each sub-tile = 16(t) x 16(n) f16 = 512B contiguous:
//   halfIndex = (tt*2+nh)*256 + ti*16 + ni   (stored as h2 pairs along n)
// ---------------------------------------------------------------------------

// ---------------------------------------------------------------------------
// Kernel 2 (dominant): SiPM contraction via WMMA.
// Block = 256 threads = 8 waves; wave w owns s-tile w (16 SiPMs) and
// accumulates 8 t-tiles (128 ticks) -> A generated once per K-chunk,
// reused by 8 WMMAs. Grid: x=ceil(550/128)=5, y=ceil(2209/128)=18, z=B.
// ---------------------------------------------------------------------------
__global__ __launch_bounds__(256) void sipm_kernel(
    const float* __restrict__ electrons,
    const float* __restrict__ dw,
    const float* __restrict__ amp_ws,
    const float* __restrict__ si_scale,
    const float* __restrict__ el_distance,
    float* __restrict__ out_sipm)
{
    __shared__ float sX[32], sY[32], sZ[32], sDW[32], sAmp[32];
    __shared__ h2    sE[2048];           // 16 sub-tiles x 256 halves (8KB)

    const int b      = blockIdx.z;
    const int tBase0 = blockIdx.x * 128;
    const int tid    = threadIdx.x;
    const int wave   = tid >> 5;
    const int lane   = tid & 31;
    const int hi     = lane >> 4;          // K-half selector (A layout)
    const int rowM   = lane & 15;          // A row / D column index
    const int sTile  = (blockIdx.y * 8 + wave) * 16;

    const int   sIdx   = sTile + rowM;
    const bool  sValid = (sIdx < S_TOT);
    const float lx     = -235.0f + 10.0f * (float)(sValid ? (sIdx / N_SIPM) : 0);
    const float ly     = -235.0f + 10.0f * (float)(sValid ? (sIdx % N_SIPM) : 0);
    const float sscale = sValid ? si_scale[sIdx] : 0.0f;
    const float el2    = el_distance[0] * el_distance[0];

    // per-lane base address into sE for the b128 loads (byte addr)
    const unsigned trBase = (unsigned)(uintptr_t)&sE[0] + (unsigned)((lane & 15) * 32 + (lane >> 4) * 16);

    v8f acc[8] = {};

    for (int n0 = 0; n0 < N_ELEC; n0 += 32) {
        __syncthreads();
        if (tid < 32) {
            const int n = n0 + tid;
            if (n < N_ELEC) {
                const float* e = electrons + ((size_t)b * N_ELEC + n) * 3;
                sX[tid] = e[0];  sY[tid] = e[1];  sZ[tid] = e[2];
                sDW[tid]  = dw[b * N_ELEC + n];
                sAmp[tid] = amp_ws[b * N_ELEC + n];
            } else {
                sX[tid] = 0.f; sY[tid] = 0.f; sZ[tid] = 1.0e9f;
                sDW[tid] = 0.f; sAmp[tid] = 0.f;
            }
            if (n0 + 32 + tid < N_ELEC)
                __builtin_prefetch(electrons + ((size_t)b * N_ELEC + n0 + 32 + tid) * 3, 0, 0);
        }
        __syncthreads();

        // E tile: 32 n x 128 t; 8 packed-pair stores per thread
        #pragma unroll
        for (int r = 0; r < 8; ++r) {
            const int q    = tid + 256 * r;
            const int niP  = q & 7;          // n pair within sub-tile
            const int ti   = (q >> 3) & 15;  // t within sub-tile
            const int nh   = (q >> 7) & 1;   // K half
            const int tt   = q >> 8;         // t tile
            const int nl   = nh * 16 + niP * 2;
            const float tv = (float)(tBase0 + tt * 16 + ti) + 0.5f;
            const float d0 = tv - sZ[nl];
            const float d1 = tv - sZ[nl + 1];
            h2 p;
            p[0] = (_Float16)(sDW[nl]     * GAUSS_NORM * __expf(-d0 * d0 * INV_2SIG));
            p[1] = (_Float16)(sDW[nl + 1] * GAUSS_NORM * __expf(-d1 * d1 * INV_2SIG));
            sE[(tt * 2 + nh) * 128 + ti * 8 + niP] = p;
        }
        __syncthreads();

        // A = R^T (16 s x 32 n), generated once, reused by 8 WMMAs
        v16h a;
        #pragma unroll
        for (int i = 0; i < 16; ++i) {
            const int v = i >> 1, w = i & 1;
            const int k = (v < 4) ? (v * 2 + w + hi * 8)
                                  : (16 + (v - 4) * 2 + w + hi * 8);
            const float dx = sX[k] - lx;
            const float dy = sY[k] - ly;
            const float r2 = dx * dx + dy * dy;
            a[i] = (_Float16)(sscale * sAmp[k] * __builtin_amdgcn_rcpf(el2 + r2 + 1e-7f));
        }

        // All 16 B sub-tiles via ds_load_b128, hazard-safe: loads and
        // the dependent s_wait_dscnt live in ONE asm block.
        v8h p0, p1, p2, p3, p4, p5, p6, p7, p8, p9, p10, p11, p12, p13, p14, p15;
        asm volatile(
            "ds_load_b128 %0, %16 offset:0\n\t"
            "ds_load_b128 %1, %16 offset:512\n\t"
            "ds_load_b128 %2, %16 offset:1024\n\t"
            "ds_load_b128 %3, %16 offset:1536\n\t"
            "ds_load_b128 %4, %16 offset:2048\n\t"
            "ds_load_b128 %5, %16 offset:2560\n\t"
            "ds_load_b128 %6, %16 offset:3072\n\t"
            "ds_load_b128 %7, %16 offset:3584\n\t"
            "ds_load_b128 %8, %16 offset:4096\n\t"
            "ds_load_b128 %9, %16 offset:4608\n\t"
            "ds_load_b128 %10, %16 offset:5120\n\t"
            "ds_load_b128 %11, %16 offset:5632\n\t"
            "ds_load_b128 %12, %16 offset:6144\n\t"
            "ds_load_b128 %13, %16 offset:6656\n\t"
            "ds_load_b128 %14, %16 offset:7168\n\t"
            "ds_load_b128 %15, %16 offset:7680\n\t"
            "s_wait_dscnt 0x0"
            : "=&v"(p0), "=&v"(p1), "=&v"(p2),  "=&v"(p3),
              "=&v"(p4), "=&v"(p5), "=&v"(p6),  "=&v"(p7),
              "=&v"(p8), "=&v"(p9), "=&v"(p10), "=&v"(p11),
              "=&v"(p12), "=&v"(p13), "=&v"(p14), "=&v"(p15)
            : "v"(trBase)
            : "memory");

        v8h lo[8] = { p0, p2, p4, p6, p8, p10, p12, p14 };
        v8h hi8[8] = { p1, p3, p5, p7, p9, p11, p13, p15 };
        #pragma unroll
        for (int tt = 0; tt < 8; ++tt) {
            v16h bm = __builtin_shufflevector(lo[tt], hi8[tt],
                                              0,1,2,3,4,5,6,7,8,9,10,11,12,13,14,15);
            acc[tt] = __builtin_amdgcn_wmma_f32_16x16x32_f16(
                false, a, false, bm, (short)0, acc[tt], false, false);
        }
    }

    // D layout: VGPR v -> row M = v + 8*hi, col N = lane&15
    #pragma unroll
    for (int tt = 0; tt < 8; ++tt) {
        const int tcol = tBase0 + tt * 16 + rowM;
        #pragma unroll
        for (int v = 0; v < 8; ++v) {
            const int s = sTile + v + 8 * hi;
            if (s < S_TOT && tcol < N_TICKS)
                out_sipm[((size_t)b * S_TOT + s) * N_TICKS + tcol] = acc[tt][v];
        }
    }
}

// ---------------------------------------------------------------------------
// Kernel 3: PMT contraction via WMMA (M=12 padded to 16).
// Block = 256 threads = 8 waves; wave w owns t-tile w (128 ticks/block).
// Grid: x=ceil(550/128)=5, y=B.
// ---------------------------------------------------------------------------
__global__ __launch_bounds__(256) void pmt_kernel(
    const float* __restrict__ electrons,
    const float* __restrict__ dw,
    const _Float16* __restrict__ pmt_ws,
    float* __restrict__ out_pmt)
{
    __shared__ float sZ[32], sDW[32];
    __shared__ h2    sE[2048];

    const int b      = blockIdx.y;
    const int tBase0 = blockIdx.x * 128;
    const int tid    = threadIdx.x;
    const int wave   = tid >> 5;
    const int lane   = tid & 31;
    const int hi     = lane >> 4;
    const int rowM   = lane & 15;     // PMT channel p (valid for p < 12)

    const unsigned trBase = (unsigned)(uintptr_t)&sE[0]
                          + (unsigned)(wave * 1024 + (lane & 15) * 32 + (lane >> 4) * 16);

    v8f acc = {};

    for (int n0 = 0; n0 < N_ELEC; n0 += 32) {
        __syncthreads();
        if (tid < 32) {
            const int n = n0 + tid;
            if (n < N_ELEC) {
                sZ[tid]  = electrons[((size_t)b * N_ELEC + n) * 3 + 2];
                sDW[tid] = dw[b * N_ELEC + n];
            } else { sZ[tid] = 1.0e9f; sDW[tid] = 0.f; }
        }
        __syncthreads();

        #pragma unroll
        for (int r = 0; r < 8; ++r) {
            const int q    = tid + 256 * r;
            const int niP  = q & 7;
            const int ti   = (q >> 3) & 15;
            const int nh   = (q >> 7) & 1;
            const int tt   = q >> 8;
            const int nl   = nh * 16 + niP * 2;
            const float tv = (float)(tBase0 + tt * 16 + ti) + 0.5f;
            const float d0 = tv - sZ[nl];
            const float d1 = tv - sZ[nl + 1];
            h2 p;
            p[0] = (_Float16)(sDW[nl]     * GAUSS_NORM * __expf(-d0 * d0 * INV_2SIG));
            p[1] = (_Float16)(sDW[nl + 1] * GAUSS_NORM * __expf(-d1 * d1 * INV_2SIG));
            sE[(tt * 2 + nh) * 128 + ti * 8 + niP] = p;
        }
        __syncthreads();

        // A = pmt_resp^T (16 p x 32 n) from workspace (f16)
        v16h a;
        #pragma unroll
        for (int i = 0; i < 16; ++i) {
            const int v = i >> 1, w = i & 1;
            const int k = (v < 4) ? (v * 2 + w + hi * 8)
                                  : (16 + (v - 4) * 2 + w + hi * 8);
            const int n = n0 + k;
            a[i] = (rowM < 12 && n < N_ELEC)
                     ? pmt_ws[((size_t)b * N_ELEC + n) * 12 + rowM]
                     : (_Float16)0.0f;
        }

        v8h q0, q1;
        asm volatile(
            "ds_load_b128 %0, %2 offset:0\n\t"
            "ds_load_b128 %1, %2 offset:512\n\t"
            "s_wait_dscnt 0x0"
            : "=&v"(q0), "=&v"(q1)
            : "v"(trBase)
            : "memory");
        v16h bm = __builtin_shufflevector(q0, q1,
                                          0,1,2,3,4,5,6,7,8,9,10,11,12,13,14,15);
        acc = __builtin_amdgcn_wmma_f32_16x16x32_f16(
                  false, a, false, bm, (short)0, acc, false, false);
    }

    const int tcol = tBase0 + wave * 16 + rowM;
    #pragma unroll
    for (int v = 0; v < 8; ++v) {
        const int p = v + 8 * hi;
        if (p < 12 && tcol < N_TICKS)
            out_pmt[((size_t)b * 12 + p) * N_TICKS + tcol] = acc[v];
    }
}

// ---------------------------------------------------------------------------
extern "C" void kernel_launch(void* const* d_in, const int* in_sizes, int n_in,
                              void* d_out, int out_size, void* d_ws, size_t ws_size,
                              hipStream_t stream) {
    (void)in_sizes; (void)n_in; (void)out_size; (void)ws_size;

    const float* electrons   = (const float*)d_in[0];
    const float* dw          = (const float*)d_in[1];
    const float* pmt_w1      = (const float*)d_in[2];
    const float* pmt_b1      = (const float*)d_in[3];
    const float* pmt_w2      = (const float*)d_in[4];
    const float* pmt_b2      = (const float*)d_in[5];
    const float* pmt_w3      = (const float*)d_in[6];
    const float* pmt_b3      = (const float*)d_in[7];
    const float* pmt_w4      = (const float*)d_in[8];
    const float* pmt_b4      = (const float*)d_in[9];
    const float* si_w1       = (const float*)d_in[10];
    const float* si_b1       = (const float*)d_in[11];
    const float* si_w2       = (const float*)d_in[12];
    const float* si_b2       = (const float*)d_in[13];
    const float* el_distance = (const float*)d_in[14];
    const float* sipm_norm   = (const float*)d_in[15];
    const float* si_scale    = (const float*)d_in[16];

    float* out_pmt  = (float*)d_out;                          // [B,12,550]
    float* out_sipm = out_pmt + (size_t)BATCH * 12 * N_TICKS; // [B,2209,550]

    _Float16* pmt_ws = (_Float16*)d_ws;
    float*    amp_ws = (float*)((char*)d_ws + (size_t)BATCH * N_ELEC * 12 * sizeof(_Float16));

    mlp_kernel<<<(BATCH * N_ELEC + 255) / 256, 256, 0, stream>>>(
        electrons, pmt_w1, pmt_b1, pmt_w2, pmt_b2, pmt_w3, pmt_b3,
        pmt_w4, pmt_b4, si_w1, si_b1, si_w2, si_b2, sipm_norm,
        pmt_ws, amp_ws);

    dim3 grid2((N_TICKS + 127) / 128, (S_TOT + 127) / 128, BATCH); // 5 x 18 x 4
    sipm_kernel<<<grid2, 256, 0, stream>>>(
        electrons, dw, amp_ws, si_scale, el_distance, out_sipm);

    dim3 grid3((N_TICKS + 127) / 128, BATCH, 1);                   // 5 x 4
    pmt_kernel<<<grid3, 256, 0, stream>>>(
        electrons, dw, pmt_ws, out_pmt);
}
